// QuantumWalkEngine_12395275616277
// MI455X (gfx1250) — compile-verified, matrix operand order checked by reading, not run on hardware
//
#include <hip/hip_runtime.h>

// ---------------------------------------------------------------------------
// QuantumWalkEngine for MI455X (gfx1250, wave32, WMMA bf16 16x16x32)
// ---------------------------------------------------------------------------

#define NCELLS 65536
#define HIDD   256

typedef __attribute__((ext_vector_type(16))) __bf16 v16bf;
typedef __attribute__((ext_vector_type(8)))  float  v8f;

union BF16x16 { v16bf v; unsigned short u[16]; };
union F32x8   { v8f  v; float f[8]; };

static __device__ __forceinline__ unsigned short f2bf(float x) {
  unsigned int u = __float_as_uint(x);
  u += 0x7FFFu + ((u >> 16) & 1u);           // round-to-nearest-even
  return (unsigned short)(u >> 16);
}

static __device__ __forceinline__ v8f wmma_bf16(v16bf a, v16bf b, v8f c) {
  return __builtin_amdgcn_wmma_f32_16x16x32_bf16(false, a, false, b, (short)0, c,
                                                 false, false);
}

// A fragment (16x32 bf16, MxK) from an LDS-resident strip, row-major, `stride`
// ushorts per row. lanes 0-15: M=lane, K pairs {0..7,16..23}; lanes 16-31:
// M=lane-16, K pairs {8..15,24..31}  (ISA 7.12.2).
static __device__ __forceinline__ v16bf load_a_frag(const unsigned short* p,
                                                    int rowBase, int stride,
                                                    int k0, int lane) {
  BF16x16 f;
  const unsigned short* r = p + (rowBase + (lane & 15)) * stride + k0 + (lane >> 4) * 8;
#pragma unroll
  for (int j = 0; j < 8; ++j) {
    int kb = ((j & 4) ? 16 : 0) + (j & 3) * 2;
    f.u[2 * j]     = r[kb];
    f.u[2 * j + 1] = r[kb + 1];
  }
  return f.v;
}

// B fragment (32x16 bf16, KxN). W is the weight matrix stored row-major [N][K]
// (i.e. B = W^T). n = n0 + (lane&15); K packing mirrors A.
static __device__ __forceinline__ v16bf load_b_frag(const unsigned short* W,
                                                    int n0, int K, int k0,
                                                    int lane) {
  BF16x16 f;
  const unsigned short* r = W + (size_t)(n0 + (lane & 15)) * K + k0 + (lane >> 4) * 8;
#pragma unroll
  for (int j = 0; j < 8; ++j) {
    int kb = ((j & 4) ? 16 : 0) + (j & 3) * 2;
    f.u[2 * j]     = r[kb];
    f.u[2 * j + 1] = r[kb + 1];
  }
  return f.v;
}

static __device__ __forceinline__ float block_reduce_sum_256(float v, float* sbuf) {
  int t = threadIdx.x;
  sbuf[t] = v;
  __syncthreads();
  for (int s = 128; s > 0; s >>= 1) {
    if (t < s) sbuf[t] += sbuf[t + s];
    __syncthreads();
  }
  float r = sbuf[0];
  __syncthreads();
  return r;
}

// ---------------------------------------------------------------------------
// Small setup kernels
// ---------------------------------------------------------------------------

__global__ void qw_init(float* scal, float* fsum, float* combined) {
  int i = blockIdx.x * blockDim.x + threadIdx.x;
  if (i < 64)   scal[i] = 0.f;
  if (i < 2048) fsum[i] = 0.f;
  if (i < 256)  combined[i] = 0.f;
}

__global__ void qw_w256(float* w256, const float* __restrict__ wih) {
  int j = blockIdx.x * blockDim.x + threadIdx.x;
  if (j < 768) w256[j] = wih[j * 257 + 256];
}

// fold x into layer-1 bias:  c[j] = b1[j] + sum_d W1[j, d] * x[d]   (d < 256)
__global__ void __launch_bounds__(256)
qw_cacg(const float* __restrict__ x,
        const float* __restrict__ ea_w1, const float* __restrict__ ea_b1,
        const float* __restrict__ eg_w1, const float* __restrict__ eg_b1,
        float* ca, float* cg) {
  __shared__ float sbuf[256];
  int j = blockIdx.x;
  const float* w; const float* b; float* dst; int jj;
  if (j < 128) { jj = j;       w = ea_w1 + (size_t)jj * 512; b = ea_b1; dst = ca; }
  else         { jj = j - 128; w = eg_w1 + (size_t)jj * 512; b = eg_b1; dst = cg; }
  float s = w[threadIdx.x] * x[threadIdx.x];
  float tot = block_reduce_sum_256(s, sbuf);
  if (threadIdx.x == 0) dst[jj] = tot + b[jj];
}

// strided f32 -> bf16 weight conversion (dst row-major [rows][cols])
__global__ void qw_conv(unsigned short* dst, const float* __restrict__ src,
                        int rows, int cols, int stride, int off) {
  int total = rows * cols;
  for (int i = blockIdx.x * blockDim.x + threadIdx.x; i < total;
       i += gridDim.x * blockDim.x) {
    int r = i / cols, c = i - r * cols;
    dst[i] = f2bf(src[(size_t)r * stride + off + c]);
  }
}

// ---------------------------------------------------------------------------
// Quantum walk: pass A (coin), pass B (hypercube gather), scalar folding
// ---------------------------------------------------------------------------

__global__ void __launch_bounds__(256)
qw_passA(const float* __restrict__ ar, const float* __restrict__ ai,
         const float* __restrict__ cr, const float* __restrict__ ci,
         float* __restrict__ c1r, float* __restrict__ c1i,
         float* __restrict__ q, float* scal) {
  __shared__ float sbuf[256];
  int n2 = blockIdx.x, d = threadIdx.x;
  size_t base = (size_t)n2 * 512;
  float a0r = ar[base + d],       a0i = ai[base + d];
  float a1r = ar[base + 256 + d], a1i = ai[base + 256 + d];
  float c00r = cr[0], c01r = cr[1], c10r = cr[2], c11r = cr[3];
  float c00i = ci[0], c01i = ci[1], c10i = ci[2], c11i = ci[3];
  float k0r = c00r * a0r - c00i * a0i + c01r * a1r - c01i * a1i;
  float k0i = c00r * a0i + c00i * a0r + c01r * a1i + c01i * a1r;
  float k1r = c10r * a0r - c10i * a0i + c11r * a1r - c11i * a1i;
  float k1i = c10r * a0i + c10i * a0r + c11r * a1i + c11i * a1r;
  size_t o = (size_t)n2 * 256 + d;
  c1r[o] = k1r; c1i[o] = k1i;
  float s1 = a0r * a0r + a0i * a0i + a1r * a1r + a1i * a1i;   // raw |amps|^2
  float q0 = k0r * k0r + k0i * k0i;                           // |coined0|^2
  float s1s = block_reduce_sum_256(s1, sbuf);
  float q0s = block_reduce_sum_256(q0, sbuf);
  if (d == 0) { q[n2] = q0s; atomicAdd(&scal[0], s1s); }
}

__global__ void __launch_bounds__(256)
qw_passB(const float* __restrict__ c1r, const float* __restrict__ c1i,
         float* __restrict__ q, float* scal) {
  __shared__ float sbuf[256];
  int n = blockIdx.x, d = threadIdx.x;
  float nr = 0.f, ni = 0.f;
#pragma unroll
  for (int b = 0; b < 16; ++b) {
    size_t m = (size_t)(n ^ (1 << b)) * 256 + d;
    nr += c1r[m]; ni += c1i[m];
  }
  nr *= (1.f / 16.f); ni *= (1.f / 16.f);
  float qs = block_reduce_sum_256(nr * nr + ni * ni, sbuf);
  if (d == 0) {
    float qn = q[n] + qs;                 // q_u[n]
    q[n] = qn;
    atomicAdd(&scal[1], qn);              // U
  }
}

// probs[n] = q_u[n] * prob_scale  (folds both normalizations + epsilons)
__global__ void qw_scalars(float* scal) {
  if (threadIdx.x == 0 && blockIdx.x == 0) {
    float S1 = scal[0], U = scal[1];
    float sc1 = 1.f / (sqrtf(S1) + 1e-8f); float s1sq = sc1 * sc1;
    float S2 = U * s1sq;
    float sc2 = 1.f / (sqrtf(S2) + 1e-8f); float s2sq = sc2 * sc2;
    float denom = S2 * s2sq + 1e-8f;
    scal[2] = s1sq * s2sq / denom;
  }
}

__global__ void __launch_bounds__(256)
qw_walkfeat(const float* __restrict__ q, const float* __restrict__ w2h_w,
            const float* __restrict__ w2h_b, const float* __restrict__ scal,
            float* wf) {
  __shared__ float sbuf[256];
  int h = blockIdx.x;
  const float* row = w2h_w + (size_t)h * 65536;
  float s = 0.f;
  for (int nn = threadIdx.x; nn < 65536; nn += 256) s += q[nn] * row[nn];
  float tot = block_reduce_sum_256(s, sbuf);
  if (threadIdx.x == 0) wf[h] = tot * scal[2] + w2h_b[h];
}

__global__ void __launch_bounds__(256)
qw_hiddens(const float* __restrict__ h0, const float* __restrict__ q,
           const float* __restrict__ scal, const float* __restrict__ wf,
           float* __restrict__ hidf, unsigned short* __restrict__ hidbf) {
  int c = blockIdx.x, d = threadIdx.x;
  size_t idx = (size_t)c * 256 + d;
  float p = q[c] * scal[2];
  float h = h0[idx] * (0.9f + 0.2f * p) + wf[d] * 0.05f;
  hidf[idx]  = h;
  hidbf[idx] = f2bf(h);
}

// ---------------------------------------------------------------------------
// Fused 2-layer MLP (256 -> relu(128) -> 256), WMMA bf16.
// modeG==0: out = layer2(a path).  modeG==1: out -= g path; emit bf16 copy,
// per-row tension, tension-sum and tension-max.
// ---------------------------------------------------------------------------

__global__ void __launch_bounds__(256)
qw_mlp(const unsigned short* __restrict__ hidbf,
       const unsigned short* __restrict__ W1, const float* __restrict__ c1,
       const unsigned short* __restrict__ W2, const float* __restrict__ b2,
       float* __restrict__ outf, unsigned short* __restrict__ outbf,
       float* __restrict__ tension, float* __restrict__ scal, int modeG) {
  __shared__ __align__(16) unsigned short As[128 * 64];
  __shared__ __align__(16) unsigned short Hs[128 * 136];
  __shared__ float trow[128];
  const int tid = threadIdx.x, lane = tid & 31, wave = tid >> 5, rs = wave * 16;
  const int row0 = blockIdx.x * 128;
  const int n = lane & 15, half = lane >> 4;
  if (tid < 128) trow[tid] = 0.f;

  // ---- layer 1: [128 rows] x K=256 -> N=128, staged in 64-wide K chunks ----
  F32x8 acc1[8];
#pragma unroll
  for (int j = 0; j < 8; ++j)
#pragma unroll
    for (int i = 0; i < 8; ++i) acc1[j].f[i] = 0.f;

#pragma unroll 1
  for (int kc = 0; kc < 4; ++kc) {
    __syncthreads();
    for (int i = tid; i < 1024; i += 256) {          // 128 rows * 8 uint4
      int r = i >> 3, seg = i & 7;
      ((uint4*)As)[i] =
          ((const uint4*)(hidbf + (size_t)(row0 + r) * 256 + kc * 64))[seg];
    }
    __syncthreads();
#pragma unroll
    for (int kk = 0; kk < 2; ++kk) {
      v16bf a = load_a_frag(As, rs, 64, kk * 32, lane);
#pragma unroll
      for (int j = 0; j < 8; ++j)
        acc1[j].v = wmma_bf16(a, load_b_frag(W1, j * 16, 256, kc * 64 + kk * 32, lane),
                              acc1[j].v);
    }
  }
  // relu epilogue -> Hs (each wave writes only its own 16 rows)
#pragma unroll
  for (int j = 0; j < 8; ++j)
#pragma unroll
    for (int i = 0; i < 8; ++i) {
      int col = j * 16 + n, m = half * 8 + i;
      float v = acc1[j].f[i] + c1[col];
      Hs[(rs + m) * 136 + col] = f2bf(fmaxf(v, 0.f));
    }
  __syncthreads();

  // ---- layer 2: K=128 -> N=256 ----
  F32x8 acc2[16];
#pragma unroll
  for (int j = 0; j < 16; ++j)
#pragma unroll
    for (int i = 0; i < 8; ++i) acc2[j].f[i] = 0.f;
#pragma unroll
  for (int kk = 0; kk < 4; ++kk) {
    v16bf a = load_a_frag(Hs, rs, 136, kk * 32, lane);
#pragma unroll
    for (int j = 0; j < 16; ++j)
      acc2[j].v = wmma_bf16(a, load_b_frag(W2, j * 16, 128, kk * 32, lane), acc2[j].v);
  }

  if (!modeG) {
#pragma unroll
    for (int j = 0; j < 16; ++j)
#pragma unroll
      for (int i = 0; i < 8; ++i) {
        int col = j * 16 + n, m = half * 8 + i;
        outf[(size_t)(row0 + rs + m) * 256 + col] = acc2[j].f[i] + b2[col];
      }
  } else {
    float part[8];
#pragma unroll
    for (int i = 0; i < 8; ++i) part[i] = 0.f;
#pragma unroll
    for (int j = 0; j < 16; ++j)
#pragma unroll
      for (int i = 0; i < 8; ++i) {
        int col = j * 16 + n, m = half * 8 + i;
        size_t idx = (size_t)(row0 + rs + m) * 256 + col;
        float o = outf[idx] - (acc2[j].f[i] + b2[col]);  // out = a - g
        outf[idx]  = o;
        outbf[idx] = f2bf(o);
        part[i] += o * o;
      }
#pragma unroll
    for (int i = 0; i < 8; ++i) atomicAdd(&trow[rs + half * 8 + i], part[i]);
    __syncthreads();
    if (tid < 128) {
      float t = trow[tid] * (1.f / 256.f);
      tension[row0 + tid] = t;
      atomicAdd(&scal[3], t);                               // tension sum
      atomicMax((int*)(scal + 4), __float_as_int(t));       // t >= 0 so int max OK
    }
  }
}

// ---------------------------------------------------------------------------
// tension epilogue / softmax combine / head
// ---------------------------------------------------------------------------

__global__ void qw_post(const float* scal, float* tmean) {
  if (threadIdx.x == 0 && blockIdx.x == 0) tmean[0] = scal[3] * (1.f / 65536.f);
}

__global__ void __launch_bounds__(256)
qw_sumexp(const float* __restrict__ tension, float* scal) {
  __shared__ float sbuf[256];
  int idx = blockIdx.x * 256 + threadIdx.x;
  float tmax = __int_as_float(((const int*)scal)[4]);
  float s = block_reduce_sum_256(expf(tension[idx] - tmax), sbuf);
  if (threadIdx.x == 0) atomicAdd(&scal[5], s);
}

__global__ void __launch_bounds__(256)
qw_combine(const float* __restrict__ outf, const float* __restrict__ tension,
           const float* __restrict__ scal, float* __restrict__ combined) {
  int d = threadIdx.x;
  int r0 = blockIdx.x * 256;
  float tmax = __int_as_float(((const int*)scal)[4]);
  float inv  = 1.f / scal[5];
  float s = 0.f;
  for (int r = 0; r < 256; ++r) {
    int row = r0 + r;
    s += expf(tension[row] - tmax) * outf[(size_t)row * 256 + d];
  }
  atomicAdd(&combined[d], s * inv);
}

__global__ void __launch_bounds__(256)
qw_head(const float* __restrict__ combined, const float* __restrict__ head_w,
        const float* __restrict__ head_b, float* __restrict__ pred) {
  int i = threadIdx.x;
  float s = head_b[i];
  for (int d = 0; d < 256; ++d) s += head_w[(size_t)i * 256 + d] * combined[d];
  pred[i] = s;
}

// ---------------------------------------------------------------------------
// Fused GRU cell, WMMA bf16.  gi uses out(bf16), gh uses hiddens(bf16);
// ir+hr and iz+hz share an accumulator.  N processed in 64-col chunks.
// ---------------------------------------------------------------------------

__global__ void __launch_bounds__(256)
qw_gru(const unsigned short* __restrict__ outbf,
       const unsigned short* __restrict__ hidbf,
       const float* __restrict__ hidf,
       const unsigned short* __restrict__ Wih,
       const unsigned short* __restrict__ Whh,
       const float* __restrict__ w256,
       const float* __restrict__ bih, const float* __restrict__ bhh,
       const float* __restrict__ tension,
       float* __restrict__ newh, float* __restrict__ fsum) {
  __shared__ __align__(16) unsigned short Ao[128 * 64];
  __shared__ __align__(16) unsigned short Ah[128 * 64];
  __shared__ float fpart[256];
  const int tid = threadIdx.x, lane = tid & 31, wave = tid >> 5, rs = wave * 16;
  const int row0 = blockIdx.x * 128;
  const int n = lane & 15, half = lane >> 4;
  fpart[tid] = 0.f;

  float tm[8];
#pragma unroll
  for (int i = 0; i < 8; ++i) tm[i] = tension[row0 + rs + half * 8 + i];

#pragma unroll 1
  for (int jc = 0; jc < 4; ++jc) {
    const int colBase = jc * 64;
    F32x8 R[4], Z[4];
#pragma unroll
    for (int j4 = 0; j4 < 4; ++j4)
#pragma unroll
      for (int i = 0; i < 8; ++i) { R[j4].f[i] = 0.f; Z[j4].f[i] = 0.f; }

    // ---- sweep 1: r and z gates over K=256 ----
#pragma unroll 1
    for (int kc = 0; kc < 4; ++kc) {
      __syncthreads();
      for (int i = tid; i < 2048; i += 256) {
        int which = i >> 10, r = (i & 1023) >> 3, seg = i & 7;
        const unsigned short* src = which ? hidbf : outbf;
        uint4 val = ((const uint4*)(src + (size_t)(row0 + r) * 256 + kc * 64))[seg];
        ((uint4*)(which ? Ah : Ao))[(r << 3) + seg] = val;
      }
      __syncthreads();
#pragma unroll
      for (int kk = 0; kk < 2; ++kk) {
        int kg = kc * 64 + kk * 32;
        v16bf ao = load_a_frag(Ao, rs, 64, kk * 32, lane);
        v16bf ah = load_a_frag(Ah, rs, 64, kk * 32, lane);
#pragma unroll
        for (int j4 = 0; j4 < 4; ++j4) {
          int nr = colBase + j4 * 16;
          R[j4].v = wmma_bf16(ao, load_b_frag(Wih, nr,       256, kg, lane), R[j4].v);
          R[j4].v = wmma_bf16(ah, load_b_frag(Whh, nr,       256, kg, lane), R[j4].v);
          Z[j4].v = wmma_bf16(ao, load_b_frag(Wih, 256 + nr, 256, kg, lane), Z[j4].v);
          Z[j4].v = wmma_bf16(ah, load_b_frag(Whh, 256 + nr, 256, kg, lane), Z[j4].v);
        }
      }
    }
#pragma unroll
    for (int j4 = 0; j4 < 4; ++j4)
#pragma unroll
      for (int i = 0; i < 8; ++i) {
        int col = colBase + j4 * 16 + n;
        float vr = R[j4].f[i] + bih[col] + bhh[col] + tm[i] * w256[col];
        R[j4].f[i] = 1.f / (1.f + expf(-vr));
        float vz = Z[j4].f[i] + bih[256 + col] + bhh[256 + col] + tm[i] * w256[256 + col];
        Z[j4].f[i] = 1.f / (1.f + expf(-vz));
      }

    // ---- sweep 2: inn / hn ----
    F32x8 NI[4], NH[4];
#pragma unroll
    for (int j4 = 0; j4 < 4; ++j4)
#pragma unroll
      for (int i = 0; i < 8; ++i) { NI[j4].f[i] = 0.f; NH[j4].f[i] = 0.f; }
#pragma unroll 1
    for (int kc = 0; kc < 4; ++kc) {
      __syncthreads();
      for (int i = tid; i < 2048; i += 256) {
        int which = i >> 10, r = (i & 1023) >> 3, seg = i & 7;
        const unsigned short* src = which ? hidbf : outbf;
        uint4 val = ((const uint4*)(src + (size_t)(row0 + r) * 256 + kc * 64))[seg];
        ((uint4*)(which ? Ah : Ao))[(r << 3) + seg] = val;
      }
      __syncthreads();
#pragma unroll
      for (int kk = 0; kk < 2; ++kk) {
        int kg = kc * 64 + kk * 32;
        v16bf ao = load_a_frag(Ao, rs, 64, kk * 32, lane);
        v16bf ah = load_a_frag(Ah, rs, 64, kk * 32, lane);
#pragma unroll
        for (int j4 = 0; j4 < 4; ++j4) {
          int nr = colBase + j4 * 16;
          NI[j4].v = wmma_bf16(ao, load_b_frag(Wih, 512 + nr, 256, kg, lane), NI[j4].v);
          NH[j4].v = wmma_bf16(ah, load_b_frag(Whh, 512 + nr, 256, kg, lane), NH[j4].v);
        }
      }
    }
    // ---- gate combine + write ----
#pragma unroll
    for (int j4 = 0; j4 < 4; ++j4)
#pragma unroll
      for (int i = 0; i < 8; ++i) {
        int col = colBase + j4 * 16 + n;
        int grow = row0 + rs + half * 8 + i;
        float inn = NI[j4].f[i] + bih[512 + col] + tm[i] * w256[512 + col];
        float hn  = NH[j4].f[i] + bhh[512 + col];
        float nc  = tanhf(inn + R[j4].f[i] * hn);
        float z   = Z[j4].f[i];
        float hp  = hidf[(size_t)grow * 256 + col];
        float h   = (1.f - z) * nc + z * hp;
        newh[(size_t)grow * 256 + col] = h;
        atomicAdd(&fpart[col], h);
      }
  }
  __syncthreads();
  int f = row0 >> 13;                      // 8192 rows per faction
  atomicAdd(&fsum[f * 256 + tid], fpart[tid]);
}

// ---------------------------------------------------------------------------
// faction sync
// ---------------------------------------------------------------------------

__global__ void qw_gprep(const float* __restrict__ fsum, float* fmean, float* gvec) {
  int c = threadIdx.x;
  float s = 0.f;
  for (int f = 0; f < 8; ++f) {
    float v = fsum[f * 256 + c];
    s += v;
    fmean[f * 256 + c] = v * (1.f / 8192.f);
  }
  gvec[c] = s * (1.f / 65536.f);           // blend preserves the mean
}

__global__ void __launch_bounds__(256)
qw_faction(float* __restrict__ newh, const float* __restrict__ fmean,
           const float* __restrict__ gvec, const int* __restrict__ step) {
  int row = blockIdx.x, c = threadIdx.x;
  size_t idx = (size_t)row * 256 + c;
  float h = newh[idx];
  int f = row >> 13;
  h = 0.85f * h + 0.15f * fmean[f * 256 + c];
  if (step[0] > 5 && (row & 8191) < 2048)
    h = 0.85f * h + 0.15f * gvec[c];
  newh[idx] = h;
}

// ---------------------------------------------------------------------------
// launcher
// ---------------------------------------------------------------------------

extern "C" void kernel_launch(void* const* d_in, const int* in_sizes, int n_in,
                              void* d_out, int out_size, void* d_ws, size_t ws_size,
                              hipStream_t stream) {
  (void)in_sizes; (void)n_in; (void)out_size; (void)ws_size;

  const float* x        = (const float*)d_in[0];
  const float* hiddens0 = (const float*)d_in[1];
  const float* amp_real = (const float*)d_in[2];
  const float* amp_imag = (const float*)d_in[3];
  const float* coin_r   = (const float*)d_in[4];
  const float* coin_i   = (const float*)d_in[5];
  const float* w2h_w    = (const float*)d_in[6];
  const float* w2h_b    = (const float*)d_in[7];
  const float* ea_w1    = (const float*)d_in[8];
  const float* ea_b1    = (const float*)d_in[9];
  const float* ea_w2    = (const float*)d_in[10];
  const float* ea_b2    = (const float*)d_in[11];
  const float* eg_w1    = (const float*)d_in[12];
  const float* eg_b1    = (const float*)d_in[13];
  const float* eg_w2    = (const float*)d_in[14];
  const float* eg_b2    = (const float*)d_in[15];
  const float* gru_wih  = (const float*)d_in[16];
  const float* gru_whh  = (const float*)d_in[17];
  const float* gru_bih  = (const float*)d_in[18];
  const float* gru_bhh  = (const float*)d_in[19];
  const float* head_w   = (const float*)d_in[20];
  const float* head_b   = (const float*)d_in[21];
  const int*   step     = (const int*)d_in[22];

  char* ws = (char*)d_ws;
  size_t off = 0;
  auto alloc = [&](size_t bytes) -> char* {
    char* p = ws + off;
    off = (off + bytes + 255) & ~(size_t)255;
    return p;
  };

  float* scal     = (float*)alloc(64 * 4);
  float* q        = (float*)alloc((size_t)NCELLS * 4);
  float* wf       = (float*)alloc(256 * 4);
  float* ca       = (float*)alloc(128 * 4);
  float* cg       = (float*)alloc(128 * 4);
  float* w256     = (float*)alloc(768 * 4);
  float* fsum     = (float*)alloc(2048 * 4);
  float* fmeanv   = (float*)alloc(2048 * 4);
  float* gvec     = (float*)alloc(256 * 4);
  float* combined = (float*)alloc(256 * 4);
  float* tension  = (float*)alloc((size_t)NCELLS * 4);
  unsigned short* Wa1h = (unsigned short*)alloc(128 * 256 * 2);
  unsigned short* Wa2  = (unsigned short*)alloc(256 * 128 * 2);
  unsigned short* Wg1h = (unsigned short*)alloc(128 * 256 * 2);
  unsigned short* Wg2  = (unsigned short*)alloc(256 * 128 * 2);
  unsigned short* Wih  = (unsigned short*)alloc(768 * 256 * 2);
  unsigned short* Whh  = (unsigned short*)alloc(768 * 256 * 2);
  float* bufA = (float*)alloc((size_t)NCELLS * 256 * 4);  // coined1.re -> out f32
  float* bufB = (float*)alloc((size_t)NCELLS * 256 * 4);  // coined1.im -> out bf16
  float* hidf = (float*)alloc((size_t)NCELLS * 256 * 4);
  unsigned short* hidbf = (unsigned short*)alloc((size_t)NCELLS * 256 * 2);

  float* c1r = bufA;
  float* c1i = bufB;
  float* outf = bufA;
  unsigned short* outbf = (unsigned short*)bufB;

  float* pred  = (float*)d_out;       // [0..255]
  float* tmean = pred + 256;          // [256]
  float* newh  = pred + 257;          // [257 .. 257 + 65536*256)

  qw_init<<<8, 256, 0, stream>>>(scal, fsum, combined);
  qw_w256<<<3, 256, 0, stream>>>(w256, gru_wih);
  qw_cacg<<<256, 256, 0, stream>>>(x, ea_w1, ea_b1, eg_w1, eg_b1, ca, cg);
  qw_conv<<<256, 256, 0, stream>>>(Wa1h, ea_w1, 128, 256, 512, 256);
  qw_conv<<<256, 256, 0, stream>>>(Wa2,  ea_w2, 256, 128, 128, 0);
  qw_conv<<<256, 256, 0, stream>>>(Wg1h, eg_w1, 128, 256, 512, 256);
  qw_conv<<<256, 256, 0, stream>>>(Wg2,  eg_w2, 256, 128, 128, 0);
  qw_conv<<<256, 256, 0, stream>>>(Wih,  gru_wih, 768, 256, 257, 0);
  qw_conv<<<256, 256, 0, stream>>>(Whh,  gru_whh, 768, 256, 256, 0);

  qw_passA<<<NCELLS, 256, 0, stream>>>(amp_real, amp_imag, coin_r, coin_i,
                                       c1r, c1i, q, scal);
  qw_passB<<<NCELLS, 256, 0, stream>>>(c1r, c1i, q, scal);
  qw_scalars<<<1, 1, 0, stream>>>(scal);
  qw_walkfeat<<<256, 256, 0, stream>>>(q, w2h_w, w2h_b, scal, wf);
  qw_hiddens<<<NCELLS, 256, 0, stream>>>(hiddens0, q, scal, wf, hidf, hidbf);

  qw_mlp<<<512, 256, 0, stream>>>(hidbf, Wa1h, ca, Wa2, ea_b2,
                                  outf, outbf, tension, scal, 0);
  qw_mlp<<<512, 256, 0, stream>>>(hidbf, Wg1h, cg, Wg2, eg_b2,
                                  outf, outbf, tension, scal, 1);

  qw_post<<<1, 1, 0, stream>>>(scal, tmean);
  qw_sumexp<<<256, 256, 0, stream>>>(tension, scal);
  qw_combine<<<256, 256, 0, stream>>>(outf, tension, scal, combined);
  qw_head<<<1, 256, 0, stream>>>(combined, head_w, head_b, pred);

  qw_gru<<<512, 256, 0, stream>>>(outbf, hidbf, hidf, Wih, Whh, w256,
                                  gru_bih, gru_bhh, tension, newh, fsum);
  qw_gprep<<<1, 256, 0, stream>>>(fsum, fmeanv, gvec);
  qw_faction<<<NCELLS, 256, 0, stream>>>(newh, fmeanv, gvec, step);
}
